// LabelGraphNodeClassifier_22978075033748
// MI455X (gfx1250) — matrix operand order repaired and checked by reading, not compiled, not run on hardware
//
#include <hip/hip_runtime.h>
#include <hip/hip_bf16.h>

// ---------- types ----------
typedef __attribute__((ext_vector_type(16))) __bf16 v16bf;
typedef __attribute__((ext_vector_type(8)))  float  v8f;

union Frag { v16bf v; uint4 u[2]; };

// ---------- CDNA5 async global->LDS staging (guarded; sync fallback) ----------
#if __has_builtin(__builtin_amdgcn_global_load_async_to_lds_b128) && \
    __has_builtin(__builtin_amdgcn_s_wait_asynccnt)
#define ASYNC_OK 1
#else
#define ASYNC_OK 0
#endif

typedef int v4i __attribute__((vector_size(4 * sizeof(int))));
typedef __attribute__((address_space(1))) v4i GV4;   // global int4
typedef __attribute__((address_space(3))) v4i LV4;   // LDS int4

__device__ __forceinline__ void stage16(const unsigned short* g, unsigned short* l) {
#if ASYNC_OK
  __builtin_amdgcn_global_load_async_to_lds_b128((GV4*)g, (LV4*)l, 0, 0);
#else
  *(uint4*)l = *(const uint4*)g;
#endif
}
#if ASYNC_OK
#define WAIT_ASYNC(n) __builtin_amdgcn_s_wait_asynccnt(n)
#else
#define WAIT_ASYNC(n)
#endif

// ---------- helpers ----------
__device__ __forceinline__ float gelu_tanh(float x) {
  float x3 = x * x * x;
  float t  = tanhf(0.7978845608028654f * (x + 0.044715f * x3));
  return 0.5f * x * (1.0f + t);
}
__device__ __forceinline__ unsigned short f2bf(float f) {
  unsigned u = __float_as_uint(f);
  unsigned r = u + 0x7FFFu + ((u >> 16) & 1u);   // round-to-nearest-even
  return (unsigned short)(r >> 16);
}
// monotone float<->uint encoding for atomicMax on floats
__device__ __forceinline__ unsigned fenc(float f) {
  unsigned u = __float_as_uint(f);
  return (u & 0x80000000u) ? ~u : (u | 0x80000000u);
}
__device__ __forceinline__ float fdec(unsigned e) {
  return (e & 0x80000000u) ? __uint_as_float(e & 0x7FFFFFFFu) : __uint_as_float(~e);
}

// ---------- elementwise f32 -> bf16 convert (opt gelu, opt per-col scale) ----------
__global__ __launch_bounds__(256) void cvt_bf16_k(
    const float* __restrict__ in, unsigned short* __restrict__ out,
    long n, int do_gelu, const float* __restrict__ colscale, int ncols) {
  long i = (long)blockIdx.x * 256 + threadIdx.x;
  if (i >= n) return;
  float v = in[i];
  if (colscale) v *= colscale[(int)(i % ncols)];
  if (do_gelu) v = gelu_tanh(v);
  out[i] = f2bf(v);
}

__global__ __launch_bounds__(256) void zero_u32_k(unsigned* __restrict__ p, long n) {
  long i = (long)blockIdx.x * 256 + threadIdx.x;
  if (i < n) p[i] = 0u;
}

// ---------- fold per-head 64x64 relation matrix into a 512x512 weight ----------
// out[h*64+j, d] = sum_i R[h,i,j] * W[h*64+i, d]   (bf16 output, N x K layout)
__global__ __launch_bounds__(256) void fuse_w_k(
    const float* __restrict__ W, const float* __restrict__ R,
    unsigned short* __restrict__ out) {
  int idx = blockIdx.x * 256 + threadIdx.x;     // 512*512
  int d   = idx & 511;
  int row = idx >> 9;                            // h*64 + j
  int h   = row >> 6, j = row & 63;
  const float* Rh = R + h * 4096;
  float s = 0.f;
#pragma unroll 8
  for (int i = 0; i < 64; ++i)
    s += Rh[i * 64 + j] * W[(size_t)(h * 64 + i) * 512 + d];
  out[idx] = f2bf(s);
}
// outb[h*64+j] = sum_i b[h*64+i] * R[h,i,j]
__global__ __launch_bounds__(256) void fuse_b_k(
    const float* __restrict__ b, const float* __restrict__ R, float* __restrict__ out) {
  int row = blockIdx.x * 256 + threadIdx.x;
  if (row >= 512) return;
  int h = row >> 6, j = row & 63;
  const float* Rh = R + h * 4096;
  float s = 0.f;
#pragma unroll 8
  for (int i = 0; i < 64; ++i) s += b[h * 64 + i] * Rh[i * 64 + j];
  out[row] = s;
}

// ---------- bf16 WMMA NT-GEMM:  C[MxN] = act(A[MxK] @ B[NxK]^T + bias) ----------
// Double-buffered LDS, async global->LDS copy of tile t+1 overlapping WMMA on tile t.
#define TM 128
#define TN 64
#define TK 64
#define AST 72                    // LDS row stride in bf16 (144 B, 16B aligned, skewed)
#define BUFELEMS ((TM + TN) * AST)

__global__ __launch_bounds__(256) void gemm_bf16_nt_k(
    const unsigned short* __restrict__ A,   // M x K (row-major, K mult of 64)
    const unsigned short* __restrict__ B,   // N x K (row-major: weight rows)
    const float* __restrict__ bias,         // N or nullptr
    float* __restrict__ C, int M, int N, int K, int ldc, int act) {
  __shared__ uint4 smem[2 * BUFELEMS * 2 / 16];   // 55,296 B
  unsigned short* base = (unsigned short*)smem;

  const int tid  = threadIdx.x;
  const int lane = tid & 31, wave = tid >> 5;
  const int half = lane >> 4, l16 = lane & 15;
  const int wm = wave & 3, wn = wave >> 2;     // 4x2 wave grid -> 32x32 per wave
  const int bm = blockIdx.y * TM, bn = blockIdx.x * TN;

  v8f acc[2][2] = {};

  // stage one K-tile into LDS buffer `buf`; OOB rows clamped (never stored later)
  auto stageTile = [&](int k0, int buf) {
    unsigned short* As = base + buf * BUFELEMS;
    unsigned short* Bs = As + TM * AST;
#pragma unroll
    for (int i = 0; i < 4; ++i) {              // A: 128x64 -> 1024 chunks, 4/thread
      int ch = tid + i * 256;
      int r = ch >> 3, c8 = ch & 7;
      int gr = bm + r; if (gr >= M) gr = M - 1;
      stage16(A + (size_t)gr * K + k0 + c8 * 8, As + r * AST + c8 * 8);
    }
#pragma unroll
    for (int i = 0; i < 2; ++i) {              // B: 64x64 -> 512 chunks, 2/thread
      int ch = tid + i * 256;
      int r = ch >> 3, c8 = ch & 7;
      int gr = bn + r; if (gr >= N) gr = N - 1;
      stage16(B + (size_t)gr * K + k0 + c8 * 8, Bs + r * AST + c8 * 8);
    }
  };

  const int nk = K / TK;
  stageTile(0, 0);

  for (int t = 0; t < nk; ++t) {
    const int buf = t & 1;
    if (t + 2 < nk) {                          // keep far tile warm in L2
      int pr = bm + (tid >> 1);
      if (pr < M) __builtin_prefetch(A + (size_t)pr * K + (t + 2) * TK + (tid & 1) * 32, 0, 3);
    }
    if (t + 1 < nk) {
      stageTile((t + 1) * TK, buf ^ 1);        // async copy overlaps compute below
      WAIT_ASYNC(6);                           // tile t (6 older ops/thread) landed
    } else {
      WAIT_ASYNC(0);
    }
    __syncthreads();

    const unsigned short* As = base + buf * BUFELEMS;
    const unsigned short* Bs = As + TM * AST;
#pragma unroll
    for (int kk = 0; kk < TK; kk += 32) {
      Frag a[2], b[2];
#pragma unroll
      for (int mi = 0; mi < 2; ++mi) {
        // A 16x32 bf16 layout: lanes0-15 K{0..7,16..23}, lanes16-31 K{8..15,24..31}
        const unsigned short* p = As + (wm * 32 + mi * 16 + l16) * AST + kk + half * 8;
        a[mi].u[0] = *(const uint4*)(p);
        a[mi].u[1] = *(const uint4*)(p + 16);
      }
#pragma unroll
      for (int ni = 0; ni < 2; ++ni) {
        // B 32x16: lane=N col; lanes0-15 K{0..7,16..23}, lanes16-31 K{8..15,24..31}
        const unsigned short* p = Bs + (wn * 32 + ni * 16 + l16) * AST + kk + half * 8;
        b[ni].u[0] = *(const uint4*)(p);
        b[ni].u[1] = *(const uint4*)(p + 16);
      }
#pragma unroll
      for (int mi = 0; mi < 2; ++mi)
#pragma unroll
        for (int ni = 0; ni < 2; ++ni)
          acc[mi][ni] = __builtin_amdgcn_wmma_f32_16x16x32_bf16(
              false, a[mi].v, false, b[ni].v, (short)0, acc[mi][ni], false, false);
    }
    __syncthreads();   // all waves done with buf before it is refilled next iter
  }

  // epilogue: C/D layout -> VGPR i: lanes0-15 M=i, lanes16-31 M=i+8; N=lane%16
#pragma unroll
  for (int mi = 0; mi < 2; ++mi)
#pragma unroll
    for (int ni = 0; ni < 2; ++ni) {
      int n = bn + wn * 32 + ni * 16 + l16;
      if (n >= N) continue;
      float bv = bias ? bias[n] : 0.f;
#pragma unroll
      for (int i = 0; i < 8; ++i) {
        int m = bm + wm * 32 + mi * 16 + half * 8 + i;
        if (m < M) {
          float v = acc[mi][ni][i] + bv;
          if (act) v = gelu_tanh(v);
          C[(size_t)m * ldc + n] = v;
        }
      }
    }
}

// ---------- edge attention passes ----------
// pass1: t[e,h] = <k2[src,h,:], q[dst,h,:]> * pri[h]/8 ; atomicMax into mx
__global__ __launch_bounds__(256) void edge_score_k(
    const float* __restrict__ k2, const float* __restrict__ q,
    const int* __restrict__ src, const int* __restrict__ dst,
    const float* __restrict__ pri, float* __restrict__ t,
    unsigned* __restrict__ mx, int E) {
  int idx = blockIdx.x * 256 + threadIdx.x;
  if (idx >= E * 8) return;
  int e = idx >> 3, h = idx & 7;
  int s = src[e], d = dst[e];
  const float4* kp = (const float4*)(k2 + (size_t)s * 512 + h * 64);
  const float4* qp = (const float4*)(q  + (size_t)d * 512 + h * 64);
  float acc = 0.f;
#pragma unroll
  for (int i = 0; i < 16; ++i) {
    float4 a = kp[i], b = qp[i];
    acc += a.x * b.x + a.y * b.y + a.z * b.z + a.w * b.w;
  }
  acc *= pri[h] * 0.125f;   // 1/sqrt(DK=64)
  t[idx] = acc;
  atomicMax(mx + d * 8 + h, fenc(acc));
}
// pass2: e = exp(t - mx[dst]); den += e
__global__ __launch_bounds__(256) void edge_exp_k(
    const int* __restrict__ dst, const unsigned* __restrict__ mx,
    float* __restrict__ t, float* __restrict__ den, int E) {
  int idx = blockIdx.x * 256 + threadIdx.x;
  if (idx >= E * 8) return;
  int e = idx >> 3, h = idx & 7;
  int d = dst[e];
  float ev = __expf(t[idx] - fdec(mx[d * 8 + h]));
  t[idx] = ev;
  atomicAdd(den + d * 8 + h, ev);
}
// pass3: agg[dst,d] += v2[src,d] * e[e,h]/max(den,1e-9)
__global__ __launch_bounds__(256) void edge_scatter_k(
    const float* __restrict__ v2, const int* __restrict__ src,
    const int* __restrict__ dst, const float* __restrict__ t,
    const float* __restrict__ den, float* __restrict__ agg, long ED) {
  long idx = (long)blockIdx.x * 256 + threadIdx.x;
  if (idx >= ED) return;
  long e = idx >> 9;
  int d = (int)(idx & 511), h = d >> 6;
  int s = src[e], dd = dst[e];
  float de  = den[dd * 8 + h];
  float att = t[e * 8 + h] / fmaxf(de, 1e-9f);
  atomicAdd(agg + (size_t)dd * 512 + d, v2[(size_t)s * 512 + d] * att);
}

// ---------- residual skip + layernorm (wave32 row reduction) ----------
__global__ __launch_bounds__(256) void resid_ln_k(
    const float* __restrict__ trans, float* __restrict__ h,
    unsigned short* __restrict__ hb, const float* __restrict__ g,
    const float* __restrict__ b, const float* __restrict__ skip, int lidx, int N) {
  int row  = blockIdx.x * 8 + (threadIdx.x >> 5);
  int lane = threadIdx.x & 31;
  if (row >= N) return;
  float alpha = 1.f / (1.f + __expf(-skip[lidx]));
  const float* tr = trans + (size_t)row * 512;
  float* hr = h + (size_t)row * 512;
  float x[16], s = 0.f, s2 = 0.f;
#pragma unroll
  for (int i = 0; i < 16; ++i) {
    float v = tr[lane + i * 32] * alpha + hr[lane + i * 32] * (1.f - alpha);
    x[i] = v; s += v; s2 += v * v;
  }
#pragma unroll
  for (int o = 16; o >= 1; o >>= 1) {
    s  += __shfl_xor(s, o, 32);
    s2 += __shfl_xor(s2, o, 32);
  }
  float mu = s * (1.f / 512.f);
  float var = s2 * (1.f / 512.f) - mu * mu;
  float rstd = rsqrtf(var + 1e-5f);
#pragma unroll
  for (int i = 0; i < 16; ++i) {
    int c = lane + i * 32;
    float v = (x[i] - mu) * rstd * g[c] + b[c];
    hr[c] = v;
    hb[(size_t)row * 512 + c] = f2bf(v);
  }
}

// ---------- host orchestration ----------
extern "C" void kernel_launch(void* const* d_in, const int* in_sizes, int n_in,
                              void* d_out, int out_size, void* d_ws, size_t ws_size,
                              hipStream_t stream) {
  const float* embeddings = (const float*)d_in[0];
  const float* node_emb   = (const float*)d_in[1];
  const float* adapt_w    = (const float*)d_in[2];
  const float* adapt_b    = (const float*)d_in[3];
  const float* kw  = (const float*)d_in[4];
  const float* kb  = (const float*)d_in[5];
  const float* qw  = (const float*)d_in[6];
  const float* qb  = (const float*)d_in[7];
  const float* vw  = (const float*)d_in[8];
  const float* vb  = (const float*)d_in[9];
  const float* aw  = (const float*)d_in[10];
  const float* ab  = (const float*)d_in[11];
  const float* pri = (const float*)d_in[12];
  const float* att_r = (const float*)d_in[13];
  const float* msg_r = (const float*)d_in[14];
  const float* skip  = (const float*)d_in[15];
  const float* ln_g  = (const float*)d_in[16];
  const float* ln_b  = (const float*)d_in[17];
  const float* out_w = (const float*)d_in[18];
  const float* out_b = (const float*)d_in[19];
  const float* attn_k = (const float*)d_in[20];
  const int* src = (const int*)d_in[21];
  const int* dst = (const int*)d_in[22];

  const int D  = 512;
  const int N  = in_sizes[1] / D;        // 20000
  const int Br = in_sizes[0] / D;        // 8192
  const int E  = in_sizes[21];           // 320000
  const int NC = 10000;                  // n_classes (constant in reference)

  // workspace carve-out (256B aligned)
  char* ws = (char*)d_ws;
  size_t off = 0;
  auto carve = [&](size_t bytes) { char* p = ws + off; off += (bytes + 255) & ~(size_t)255; return p; };
  float*          h    = (float*)carve((size_t)N * D * 4);
  unsigned short* hb   = (unsigned short*)carve((size_t)N * D * 2);
  float*          k2   = (float*)carve((size_t)N * D * 4);
  float*          q    = (float*)carve((size_t)N * D * 4);
  float*          v2   = (float*)carve((size_t)N * D * 4);
  float*          agg  = (float*)carve((size_t)N * D * 4);
  unsigned short* aggb = (unsigned short*)carve((size_t)N * D * 2);
  float*          tb   = (float*)carve((size_t)E * 8 * 4);
  unsigned*       mx   = (unsigned*)carve((size_t)N * 8 * 4);
  float*          den  = (float*)carve((size_t)N * 8 * 4);
  unsigned short* wf1  = (unsigned short*)carve(512 * 512 * 2);
  unsigned short* wf2  = (unsigned short*)carve(512 * 512 * 2);
  unsigned short* wf3  = (unsigned short*)carve(512 * 512 * 2);
  float*          fb1  = (float*)carve(512 * 4);
  float*          fb3  = (float*)carve(512 * 4);
  // phase reuse
  float*          trans = k2;                       // after edge passes
  float*          cls   = q;                        // NC*D f32
  unsigned short* clsb  = (unsigned short*)v2;      // NC*D bf16
  unsigned short* embb  = (unsigned short*)agg;     // Br*D bf16 (final phase)

  auto cdiv = [](long a, long b) { return (int)((a + b - 1) / b); };
  dim3 blk(256);

  // adapt: h = gelu(node_emb @ adapt_w^T + adapt_b)
  cvt_bf16_k<<<cdiv((long)N * D, 256), blk, 0, stream>>>(node_emb, hb, (long)N * D, 0, nullptr, D);
  cvt_bf16_k<<<cdiv(512L * 512, 256), blk, 0, stream>>>(adapt_w, wf1, 512L * 512, 0, nullptr, D);
  {
    dim3 g(cdiv(D, TN), cdiv(N, TM));
    gemm_bf16_nt_k<<<g, blk, 0, stream>>>(hb, wf1, adapt_b, h, N, D, D, D, 1);
  }
  cvt_bf16_k<<<cdiv((long)N * D, 256), blk, 0, stream>>>(h, hb, (long)N * D, 0, nullptr, D);

  for (int l = 0; l < 2; ++l) {
    const float* Ra = att_r + (size_t)l * 8 * 64 * 64;
    const float* Rm = msg_r + (size_t)l * 8 * 64 * 64;
    // fold relation matrices into k/v weights
    fuse_w_k<<<1024, blk, 0, stream>>>(kw + (size_t)l * D * D, Ra, wf1);
    fuse_b_k<<<2,    blk, 0, stream>>>(kb + (size_t)l * D, Ra, fb1);
    fuse_w_k<<<1024, blk, 0, stream>>>(vw + (size_t)l * D * D, Rm, wf3);
    fuse_b_k<<<2,    blk, 0, stream>>>(vb + (size_t)l * D, Rm, fb3);
    cvt_bf16_k<<<cdiv(512L * 512, 256), blk, 0, stream>>>(qw + (size_t)l * D * D, wf2, 512L * 512, 0, nullptr, D);

    dim3 g(cdiv(D, TN), cdiv(N, TM));
    gemm_bf16_nt_k<<<g, blk, 0, stream>>>(hb, wf1, fb1, k2, N, D, D, D, 0);
    gemm_bf16_nt_k<<<g, blk, 0, stream>>>(hb, wf2, qb + (size_t)l * D, q, N, D, D, D, 0);
    gemm_bf16_nt_k<<<g, blk, 0, stream>>>(hb, wf3, fb3, v2, N, D, D, D, 0);

    zero_u32_k<<<cdiv((long)N * 8, 256), blk, 0, stream>>>(mx, (long)N * 8);
    zero_u32_k<<<cdiv((long)N * 8, 256), blk, 0, stream>>>((unsigned*)den, (long)N * 8);
    zero_u32_k<<<cdiv((long)N * D, 256), blk, 0, stream>>>((unsigned*)agg, (long)N * D);

    edge_score_k<<<cdiv((long)E * 8, 256), blk, 0, stream>>>(k2, q, src, dst, pri + l * 8, tb, mx, E);
    edge_exp_k<<<cdiv((long)E * 8, 256), blk, 0, stream>>>(dst, mx, tb, den, E);
    edge_scatter_k<<<cdiv((long)E * D, 256), blk, 0, stream>>>(v2, src, dst, tb, den, agg, (long)E * D);

    // trans = gelu(agg) @ aw^T + ab
    cvt_bf16_k<<<cdiv((long)N * D, 256), blk, 0, stream>>>(agg, aggb, (long)N * D, 1, nullptr, D);
    cvt_bf16_k<<<cdiv(512L * 512, 256), blk, 0, stream>>>(aw + (size_t)l * D * D, wf1, 512L * 512, 0, nullptr, D);
    gemm_bf16_nt_k<<<g, blk, 0, stream>>>(aggb, wf1, ab + (size_t)l * D, trans, N, D, D, D, 0);

    resid_ln_k<<<cdiv(N, 8), blk, 0, stream>>>(trans, h, hb,
        ln_g + (size_t)l * D, ln_b + (size_t)l * D, skip, l, N);
  }

  // cls_emb = h[:NC] @ out_w^T + out_b
  cvt_bf16_k<<<cdiv(512L * 512, 256), blk, 0, stream>>>(out_w, wf1, 512L * 512, 0, nullptr, D);
  {
    dim3 g(cdiv(D, TN), cdiv(NC, TM));
    gemm_bf16_nt_k<<<g, blk, 0, stream>>>(hb, wf1, out_b, cls, NC, D, D, D, 0);
  }
  cvt_bf16_k<<<cdiv((long)NC * D, 256), blk, 0, stream>>>(cls, clsb, (long)NC * D, 0, nullptr, D);

  // logits = (embeddings * attn_kernels) @ cls_emb^T
  cvt_bf16_k<<<cdiv((long)Br * D, 256), blk, 0, stream>>>(embeddings, embb, (long)Br * D, 0, attn_k, D);
  {
    dim3 g(cdiv(NC, TN), cdiv(Br, TM));
    gemm_bf16_nt_k<<<g, blk, 0, stream>>>(embb, clsb, nullptr, (float*)d_out, Br, NC, D, NC, 0);
  }
}
